// BinaryLinear_72155450573461
// MI455X (gfx1250) — matrix operand order repaired and checked by reading, not compiled, not run on hardware
//
#include <hip/hip_runtime.h>
#include <hip/hip_bf16.h>

typedef __attribute__((ext_vector_type(16))) _Float16 v16h;
typedef __attribute__((ext_vector_type(8)))  _Float16 v8h;
typedef __attribute__((ext_vector_type(8)))  float    v8f;

static __device__ __forceinline__ v16h cat8(v8h lo, v8h hi) {
  return __builtin_shufflevector(lo, hi, 0,1,2,3,4,5,6,7,8,9,10,11,12,13,14,15);
}

#define WMMA_F16(a, b, c) \
  __builtin_amdgcn_wmma_f32_16x16x32_f16(false, (a), false, (b), (short)0, (c), false, false)

// ---------------------------------------------------------------------------
// Weight binarization: Wh[n,k] = sign(W[n,k]) as f16, zero-padded to Kdst cols
// ---------------------------------------------------------------------------
__global__ void pack_sign(const float* __restrict__ W, _Float16* __restrict__ Wh,
                          int N, int Ksrc, int Kdst) {
  int idx = blockIdx.x * blockDim.x + threadIdx.x;
  if (idx >= N * Kdst) return;
  int n = idx / Kdst;
  int k = idx - n * Kdst;
  float s = 0.f;
  if (k < Ksrc) {
    float w = W[(size_t)n * Ksrc + k];
    s = (w > 0.f) ? 1.f : ((w < 0.f) ? -1.f : 0.f);
  }
  Wh[idx] = (_Float16)s;
}

// ---------------------------------------------------------------------------
// WMMA GEMM: C[M,N](f16) = A[M,K] * W[N,K]^T, W already f16 (+/-1), A f32 or f16
// Block tile 128x128, 256 threads = 8 waves (wave32), wave tile 32x64 (2x4 WMMA
// tiles), double-buffered LDS (stride 40 halves for bank-conflict relief).
// Epilogue stages C through LDS for fully coalesced b128 global stores.
// ---------------------------------------------------------------------------
template <bool AF32>
__global__ __launch_bounds__(256)
void wmma_gemm(const void* __restrict__ Aptr, const _Float16* __restrict__ Wp,
               _Float16* __restrict__ Cp, int N, int Kpad, int Kvalid, int lda) {
  __shared__ alignas(16) char smem[40960];       // A/B tiles; reused for C tile
  auto lA = (_Float16 (*)[128][40])smem;         // lA[2][128][40]
  auto lB = (_Float16 (*)[128][40])(smem + 20480);

  const int t    = threadIdx.x;
  const int lane = t & 31;
  const int wave = t >> 5;
  const int wm   = wave >> 1;        // 0..3 : 32-row band within block tile
  const int wn   = wave & 1;         // 0..1 : 64-col band within block tile

  const int rIn   = t >> 1;          // 0..127 : tile row this thread stages
  const int kHalf = (t & 1) << 4;    // 0 or 16 : which 16-wide K chunk

  const size_t gRow = (size_t)blockIdx.y * 128 + rIn;   // activation row
  const size_t wRow = (size_t)blockIdx.x * 128 + rIn;   // weight row (output col)

  const float*    Af = (const float*)Aptr;
  const _Float16* Ah = (const _Float16*)Aptr;

  v8h aS0, aS1, bS0, bS1;            // staging regs for global->LDS

  const int KT = Kpad >> 5;

  // per-lane fragment addressing (wave32; A and B use the same contiguous k order)
  const int mrow = lane & 15;                 // A row within 16x32 tile
  const int aks  = (lane < 16) ? 0 : 16;      // A: 16 contiguous k per lane
  const int bcol = lane & 15;                 // B column
  const int bkb  = (lane < 16) ? 0 : 16;      // B: 16 contiguous k per lane

  v8f acc[2][4];
  const v8f vzero = {0.f, 0.f, 0.f, 0.f, 0.f, 0.f, 0.f, 0.f};
#pragma unroll
  for (int i = 0; i < 2; ++i)
#pragma unroll
    for (int j = 0; j < 4; ++j) acc[i][j] = vzero;

  auto loadA = [&](int kt) {
    const int k0 = (kt << 5) + kHalf;
    if (AF32) {
      float f[16];
      if (k0 + 16 <= Kvalid) {
        const float* p = Af + gRow * (size_t)lda + k0;
#pragma unroll
        for (int i = 0; i < 4; ++i) {
          float4 v = *(const float4*)(p + i * 4);
          f[i * 4 + 0] = v.x; f[i * 4 + 1] = v.y;
          f[i * 4 + 2] = v.z; f[i * 4 + 3] = v.w;
        }
      } else {
#pragma unroll
        for (int i = 0; i < 16; ++i) f[i] = 0.f;
      }
#pragma unroll
      for (int i = 0; i < 8; ++i) { aS0[i] = (_Float16)f[i]; aS1[i] = (_Float16)f[8 + i]; }
    } else {
      const _Float16* p = Ah + gRow * (size_t)lda + k0;
      aS0 = *(const v8h*)p;
      aS1 = *(const v8h*)(p + 8);
    }
  };
  auto loadB = [&](int kt) {
    const int k0 = (kt << 5) + kHalf;
    const _Float16* p = Wp + wRow * (size_t)Kpad + k0;
    bS0 = *(const v8h*)p;
    bS1 = *(const v8h*)(p + 8);
  };
  auto storeStage = [&](int s) {
    *(v8h*)&lA[s][rIn][kHalf]     = aS0;
    *(v8h*)&lA[s][rIn][kHalf + 8] = aS1;
    *(v8h*)&lB[s][rIn][kHalf]     = bS0;
    *(v8h*)&lB[s][rIn][kHalf + 8] = bS1;
  };
  auto compute = [&](int s) {
    // load ALL fragments into distinct live values first so the 12 ds_load_b128
    // issue back-to-back, then the 8 WMMAs drain behind a single dscnt wait
    const _Float16* pa0 = &lA[s][wm * 32 + mrow][aks];
    const _Float16* pa1 = &lA[s][wm * 32 + 16 + mrow][aks];
    const _Float16* pb0 = &lB[s][wn * 64 +  0 + bcol][bkb];
    const _Float16* pb1 = &lB[s][wn * 64 + 16 + bcol][bkb];
    const _Float16* pb2 = &lB[s][wn * 64 + 32 + bcol][bkb];
    const _Float16* pb3 = &lB[s][wn * 64 + 48 + bcol][bkb];
    v16h a0 = cat8(*(const v8h*)pa0, *(const v8h*)(pa0 + 8));
    v16h a1 = cat8(*(const v8h*)pa1, *(const v8h*)(pa1 + 8));
    v16h b0 = cat8(*(const v8h*)pb0, *(const v8h*)(pb0 + 8));
    v16h b1 = cat8(*(const v8h*)pb1, *(const v8h*)(pb1 + 8));
    v16h b2 = cat8(*(const v8h*)pb2, *(const v8h*)(pb2 + 8));
    v16h b3 = cat8(*(const v8h*)pb3, *(const v8h*)(pb3 + 8));
    acc[0][0] = WMMA_F16(a0, b0, acc[0][0]);
    acc[1][0] = WMMA_F16(a1, b0, acc[1][0]);
    acc[0][1] = WMMA_F16(a0, b1, acc[0][1]);
    acc[1][1] = WMMA_F16(a1, b1, acc[1][1]);
    acc[0][2] = WMMA_F16(a0, b2, acc[0][2]);
    acc[1][2] = WMMA_F16(a1, b2, acc[1][2]);
    acc[0][3] = WMMA_F16(a0, b3, acc[0][3]);
    acc[1][3] = WMMA_F16(a1, b3, acc[1][3]);
  };

  loadA(0); loadB(0); storeStage(0);
  __syncthreads();
  for (int kt = 0; kt < KT; ++kt) {
    const int cur = kt & 1;
    const bool more = (kt + 1) < KT;
    if (more) { loadA(kt + 1); loadB(kt + 1); }
    compute(cur);
    if (more) { storeStage(cur ^ 1); __syncthreads(); }
  }

  // ---- epilogue: stage C tile in LDS, then coalesced b128 stores ----------
  __syncthreads();                               // all waves done reading lA/lB
  auto lC = (_Float16 (*)[136])smem;             // 128 x 136 halves = 34816 B
  const int rowHi = (lane >> 4) << 3;            // C/D: VGPR v -> row v + rowHi
#pragma unroll
  for (int i = 0; i < 2; ++i) {
#pragma unroll
    for (int j = 0; j < 4; ++j) {
      const int r0 = wm * 32 + i * 16 + rowHi;
      const int c  = wn * 64 + j * 16 + bcol;
#pragma unroll
      for (int v = 0; v < 8; ++v)
        lC[r0 + v][c] = (_Float16)acc[i][j][v];
    }
  }
  __syncthreads();
  {
    const int row = t >> 1;
    const int ch  = (t & 1) * 64;
    const size_t gr = (size_t)blockIdx.y * 128 + row;
    _Float16* dst = Cp + gr * (size_t)N + (size_t)blockIdx.x * 128 + ch;
    const _Float16* src = &lC[row][ch];
#pragma unroll
    for (int q = 0; q < 8; ++q)
      *(v8h*)(dst + q * 8) = *(const v8h*)(src + q * 8);
  }
}

// ---------------------------------------------------------------------------
// Per-column sum / sum-of-squares (optionally of relu(x)) via f32 atomics
// ---------------------------------------------------------------------------
__global__ void col_stats(const _Float16* __restrict__ X, float* __restrict__ sum,
                          float* __restrict__ sq, int cols, int rowsPer, int reluFirst) {
  const int c  = blockIdx.x * blockDim.x + threadIdx.x;
  const int r0 = blockIdx.y * rowsPer;
  float s = 0.f, q = 0.f;
  for (int r = r0; r < r0 + rowsPer; ++r) {
    float v = (float)X[(size_t)r * cols + c];
    if (reluFirst) v = fmaxf(v, 0.f);
    s += v; q += v * v;
  }
  atomicAdd(&sum[c], s);
  atomicAdd(&sq[c], q);
}

__global__ void bn_finalize(const float* __restrict__ sum, const float* __restrict__ sq,
                            float* __restrict__ mu, float* __restrict__ is,
                            int cols, float invB) {
  int c = blockIdx.x * blockDim.x + threadIdx.x;
  if (c >= cols) return;
  float m   = sum[c] * invB;
  float var = sq[c] * invB - m * m;
  mu[c] = m;
  is[c] = rsqrtf(var + 1e-5f);
}

// reluFirst=0: Y = relu(bn(X));  reluFirst=1: Y = bn(relu(X))
__global__ void bn_apply(const _Float16* __restrict__ X, _Float16* __restrict__ Y,
                         const float* __restrict__ mu, const float* __restrict__ is,
                         const float* __restrict__ g, const float* __restrict__ b,
                         int cols, size_t total, int reluFirst) {
  size_t i = (size_t)blockIdx.x * blockDim.x + threadIdx.x;
  if (i >= total) return;
  int c = (int)(i % cols);
  float v = (float)X[i];
  float y;
  if (reluFirst) {
    v = fmaxf(v, 0.f);
    y = g[c] * (v - mu[c]) * is[c] + b[c];
  } else {
    y = fmaxf(g[c] * (v - mu[c]) * is[c] + b[c], 0.f);
  }
  Y[i] = (_Float16)y;
}

// ---------------------------------------------------------------------------
// Head: logits = relu(A[row,:] . signW3[j,:]) for j<10, then softmax -> f32 out
// One wave32 per row; W3 staged in LDS.
// ---------------------------------------------------------------------------
__global__ __launch_bounds__(256)
void head_softmax(const _Float16* __restrict__ A, const _Float16* __restrict__ W3,
                  float* __restrict__ out, int K) {
  __shared__ _Float16 w[10 * 1024];
  for (int i = threadIdx.x; i < 10 * 1024; i += 256) w[i] = W3[i];
  __syncthreads();

  const int lane = threadIdx.x & 31;
  const int wv   = threadIdx.x >> 5;
  const int row  = blockIdx.x * 8 + wv;

  float acc[10];
#pragma unroll
  for (int j = 0; j < 10; ++j) acc[j] = 0.f;

  const _Float16* a = A + (size_t)row * K;
  for (int k = lane; k < K; k += 32) {
    float av = (float)a[k];
#pragma unroll
    for (int j = 0; j < 10; ++j) acc[j] += av * (float)w[j * 1024 + k];
  }
#pragma unroll
  for (int j = 0; j < 10; ++j)
    for (int off = 16; off; off >>= 1) acc[j] += __shfl_xor(acc[j], off, 32);

  if (lane == 0) {
    float l[10], m = 0.f;
#pragma unroll
    for (int j = 0; j < 10; ++j) { l[j] = fmaxf(acc[j], 0.f); m = fmaxf(m, l[j]); }
    float s = 0.f;
#pragma unroll
    for (int j = 0; j < 10; ++j) { l[j] = expf(l[j] - m); s += l[j]; }
    float inv = 1.f / s;
#pragma unroll
    for (int j = 0; j < 10; ++j) out[(size_t)row * 10 + j] = l[j] * inv;
  }
}

// ---------------------------------------------------------------------------
extern "C" void kernel_launch(void* const* d_in, const int* in_sizes, int n_in,
                              void* d_out, int out_size, void* d_ws, size_t ws_size,
                              hipStream_t stream) {
  (void)in_sizes; (void)n_in; (void)out_size; (void)ws_size;

  const int B = 32768, D = 784, Dpad = 800, H = 1024, C = 10;

  const float* x  = (const float*)d_in[0];
  const float* W1 = (const float*)d_in[1];
  const float* g1 = (const float*)d_in[2];
  const float* b1 = (const float*)d_in[3];
  const float* W2 = (const float*)d_in[4];
  const float* g2 = (const float*)d_in[5];
  const float* b2 = (const float*)d_in[6];
  const float* W3 = (const float*)d_in[7];
  float* out = (float*)d_out;

  // workspace layout (256B-aligned offsets)
  char* w = (char*)d_ws;
  _Float16* W1h = (_Float16*)(w);                      // 1024*800*2   = 1,638,400
  _Float16* W2h = (_Float16*)(w + 1638400);            // 1024*1024*2  = 2,097,152
  _Float16* W3h = (_Float16*)(w + 3735552);            // 10*1024*2    (pad to 20,480)
  float*    st  = (float*)   (w + 3756032);            // 8*1024 f32 stats
  _Float16* H1  = (_Float16*)(w + 3788800);            // 32768*1024*2 = 67,108,864
  _Float16* A2  = (_Float16*)(w + 70897664);           // 32768*1024*2 = 67,108,864
  _Float16* G2  = H1;   // reuse: H1 consumed before GEMM2 writes
  _Float16* A3  = A2;   // reuse: A2 consumed before apply2 writes

  float* sums1 = st;          float* sqs1 = st + 1024;
  float* sums2 = st + 2048;   float* sqs2 = st + 3072;
  float* mu1   = st + 4096;   float* is1  = st + 5120;
  float* mu2   = st + 6144;   float* is2  = st + 7168;

  // zero the accumulation stats (ws is poisoned, atomics need 0 start)
  hipMemsetAsync(st, 0, 4096 * sizeof(float), stream);

  // binarize weights
  pack_sign<<<(H * Dpad + 255) / 256, 256, 0, stream>>>(W1, W1h, H, D, Dpad);
  pack_sign<<<(H * H + 255) / 256, 256, 0, stream>>>(W2, W2h, H, H, H);
  pack_sign<<<(C * H + 255) / 256, 256, 0, stream>>>(W3, W3h, C, H, H);

  const size_t total = (size_t)B * H;
  dim3 gGemm(H / 128, B / 128);

  // layer 1: H1 = x @ signW1^T   (x converted f32->f16 while staging to LDS)
  wmma_gemm<true><<<gGemm, 256, 0, stream>>>(x, W1h, H1, H, Dpad, D, D);
  col_stats<<<dim3(H / 256, B / 512), 256, 0, stream>>>(H1, sums1, sqs1, H, 512, 0);
  bn_finalize<<<H / 256, 256, 0, stream>>>(sums1, sqs1, mu1, is1, H, 1.f / B);
  bn_apply<<<(unsigned)(total / 256), 256, 0, stream>>>(H1, A2, mu1, is1, g1, b1, H, total, 0);

  // layer 2: G2 = A2 @ signW2^T ; A3 = bn(relu(G2))
  wmma_gemm<false><<<gGemm, 256, 0, stream>>>(A2, W2h, G2, H, H, H, H);
  col_stats<<<dim3(H / 256, B / 512), 256, 0, stream>>>(G2, sums2, sqs2, H, 512, 1);
  bn_finalize<<<H / 256, 256, 0, stream>>>(sums2, sqs2, mu2, is2, H, 1.f / B);
  bn_apply<<<(unsigned)(total / 256), 256, 0, stream>>>(G2, A3, mu2, is2, g2, b2, H, total, 1);

  // head: relu(A3 @ signW3^T) + softmax
  head_softmax<<<B / 8, 256, 0, stream>>>(A3, W3h, out, H);
}
